// LRML_90804198572513
// MI455X (gfx1250) — compile-verified, matrix operand order checked by reading, not run on hardware
//
#include <hip/hip_runtime.h>
#include <math.h>

typedef __attribute__((ext_vector_type(16))) _Float16 v16h;
typedef __attribute__((ext_vector_type(8)))  float    v8f;

#define WAVES_PER_BLOCK 8
#define ROWS_PER_WAVE   16
#define ROWS_PER_BLOCK  (WAVES_PER_BLOCK * ROWS_PER_WAVE)
#define EMB_D           32
#define M_ATT           10
#define SC_STRIDE       20   // 80B rows: 16B-aligned, conflict-light
#define RL_STRIDE       36   // 144B rows: 16B-aligned

__launch_bounds__(256)
__global__ void lrml_fused_wmma(const int* __restrict__ user_ids,
                                const int* __restrict__ item_ids,
                                const float* __restrict__ user_emb,
                                const float* __restrict__ item_emb,
                                const float* __restrict__ W_att,
                                const float* __restrict__ mem_tbl,
                                float* __restrict__ out,
                                int B)
{
    // Per-wave LDS tiles used only for fragment-layout changes
    // (C/D layout <-> row-major). Same-wave DS ops are in-order on CDNA5,
    // so no barriers are required anywhere.
    __shared__ __align__(16) float ssc [WAVES_PER_BLOCK][ROWS_PER_WAVE][SC_STRIDE];
    __shared__ __align__(16) float srel[WAVES_PER_BLOCK][ROWS_PER_WAVE][RL_STRIDE];

    const int lane  = threadIdx.x & 31;
    const int wave  = threadIdx.x >> 5;
    const int grp   = lane >> 4;    // 0: lanes 0-15, 1: lanes 16-31
    const int lrow  = lane & 15;    // A-row / B-col / C-col index
    const int kbase = grp << 3;     // K half base: 0 or 8

    const int rowbase = (blockIdx.x * WAVES_PER_BLOCK + wave) * ROWS_PER_WAVE;
    const int grow = rowbase + lrow;
    const int gidx = (grow < B) ? grow : (B - 1);

    // ---- Gather ue / ie rows. Lane l holds row (l&15), K in
    // {kbase..kbase+7} U {kbase+16..kbase+23}: exactly the f16 A-fragment
    // striping, fetched as four 128-bit loads per table. ----
    const long long uid = (long long)user_ids[gidx];
    const long long iid = (long long)item_ids[gidx];
    const float* up = user_emb + uid * EMB_D;
    const float* ip = item_emb + iid * EMB_D;

    float ur[16], ir[16];
    {
        float4 t;
        t = *(const float4*)(up + kbase);      ur[0]=t.x; ur[1]=t.y; ur[2]=t.z; ur[3]=t.w;
        t = *(const float4*)(up + kbase + 4);  ur[4]=t.x; ur[5]=t.y; ur[6]=t.z; ur[7]=t.w;
        t = *(const float4*)(up + kbase + 16); ur[8]=t.x; ur[9]=t.y; ur[10]=t.z; ur[11]=t.w;
        t = *(const float4*)(up + kbase + 20); ur[12]=t.x; ur[13]=t.y; ur[14]=t.z; ur[15]=t.w;
        t = *(const float4*)(ip + kbase);      ir[0]=t.x; ir[1]=t.y; ir[2]=t.z; ir[3]=t.w;
        t = *(const float4*)(ip + kbase + 4);  ir[4]=t.x; ir[5]=t.y; ir[6]=t.z; ir[7]=t.w;
        t = *(const float4*)(ip + kbase + 16); ir[8]=t.x; ir[9]=t.y; ir[10]=t.z; ir[11]=t.w;
        t = *(const float4*)(ip + kbase + 20); ir[12]=t.x; ir[13]=t.y; ir[14]=t.z; ir[15]=t.w;
    }

    // ---- Renorm to max_norm = 1: e *= 1/max(||e||, 1). Partner lane
    // (xor 16) holds the other half of the row. ----
    float su = 0.f, si = 0.f;
#pragma unroll
    for (int i = 0; i < 16; ++i) { su += ur[i] * ur[i]; si += ir[i] * ir[i]; }
    su += __shfl_xor(su, 16, 32);
    si += __shfl_xor(si, 16, 32);
    const float us = 1.0f / fmaxf(sqrtf(su), 1.0f);
    const float is = 1.0f / fmaxf(sqrtf(si), 1.0f);
#pragma unroll
    for (int i = 0; i < 16; ++i) { ur[i] *= us; ir[i] *= is; }

    // ---- A fragment: joint = ue * ie (f16), already in fragment order. ----
    v16h aj;
#pragma unroll
    for (int e = 0; e < 16; ++e) aj[e] = (_Float16)(ur[e] * ir[e]);

    // ---- B fragment: W_att^T [32 x 16], column n = lrow (zero-pad n>=10). ----
    v16h bw;
    if (lrow < M_ATT) {
        const float* wp = W_att + lrow * EMB_D;
        float4 w0 = *(const float4*)(wp + kbase);
        float4 w1 = *(const float4*)(wp + kbase + 4);
        float4 w2 = *(const float4*)(wp + kbase + 16);
        float4 w3 = *(const float4*)(wp + kbase + 20);
        bw[0]=(_Float16)w0.x;  bw[1]=(_Float16)w0.y;  bw[2]=(_Float16)w0.z;  bw[3]=(_Float16)w0.w;
        bw[4]=(_Float16)w1.x;  bw[5]=(_Float16)w1.y;  bw[6]=(_Float16)w1.z;  bw[7]=(_Float16)w1.w;
        bw[8]=(_Float16)w2.x;  bw[9]=(_Float16)w2.y;  bw[10]=(_Float16)w2.z; bw[11]=(_Float16)w2.w;
        bw[12]=(_Float16)w3.x; bw[13]=(_Float16)w3.y; bw[14]=(_Float16)w3.z; bw[15]=(_Float16)w3.w;
    } else {
#pragma unroll
        for (int e = 0; e < 16; ++e) bw[e] = (_Float16)0.0f;
    }

    // ---- GEMM1: logits[16x16] = joint[16x32] x W_att^T[32x16] ----
    v8f zacc = {};
    v8f logits = __builtin_amdgcn_wmma_f32_16x16x32_f16(
        false, aj, false, bw, (short)0, zacc, false, false);

    // ---- C-layout -> row-major spill of logits (8 ds_store_b32). ----
#pragma unroll
    for (int j = 0; j < 8; ++j)
        ssc[wave][j + 8 * grp][lrow] = logits[j];

    // ---- Serial per-lane softmax over the lane's own row (lrow).
    // Lanes l and l+16 redundantly process the same row; each directly
    // produces its half of the GEMM2 A-fragment. No shuffles. ----
    const float* srow = &ssc[wave][lrow][0];
    float4 x0 = *(const float4*)(srow + 0);
    float4 x1 = *(const float4*)(srow + 4);
    float4 x2 = *(const float4*)(srow + 8);
    float r[10] = { x0.x, x0.y, x0.z, x0.w, x1.x, x1.y, x1.z, x1.w, x2.x, x2.y };
    float mx = r[0];
#pragma unroll
    for (int n = 1; n < M_ATT; ++n) mx = fmaxf(mx, r[n]);
    float ex[M_ATT];
    float denom = 0.f;
#pragma unroll
    for (int n = 0; n < M_ATT; ++n) { ex[n] = __expf(r[n] - mx); denom += ex[n]; }
    const float inv = 1.0f / denom;

    v16h as;
#pragma unroll
    for (int e = 0; e < 16; ++e) as[e] = (_Float16)0.0f;
    if (grp == 0) {
#pragma unroll
        for (int e = 0; e < 8; ++e) as[e] = (_Float16)(ex[e] * inv);  // K = 0..7
    } else {
        as[0] = (_Float16)(ex[8] * inv);                              // K = 8
        as[1] = (_Float16)(ex[9] * inv);                              // K = 9
    }

    // ---- B fragments: memory[10x32] zero-padded to [32x16] per N-tile. ----
    v16h bm0, bm1;
#pragma unroll
    for (int e = 0; e < 16; ++e) {
        const int K = kbase + ((e < 8) ? e : e + 8);
        float m0 = 0.0f, m1 = 0.0f;
        if (K < M_ATT) {
            m0 = mem_tbl[K * EMB_D + lrow];
            m1 = mem_tbl[K * EMB_D + 16 + lrow];
        }
        bm0[e] = (_Float16)m0;
        bm1[e] = (_Float16)m1;
    }

    // ---- GEMM2: rel tiles [16x16] each (d = 0..15 and 16..31). ----
    v8f rel0 = __builtin_amdgcn_wmma_f32_16x16x32_f16(
        false, as, false, bm0, (short)0, zacc, false, false);
    v8f rel1 = __builtin_amdgcn_wmma_f32_16x16x32_f16(
        false, as, false, bm1, (short)0, zacc, false, false);

    // ---- C-layout -> A-layout for rel via LDS, then the squared distance
    // is computed per lane against ur/ir that are already in A order. ----
#pragma unroll
    for (int j = 0; j < 8; ++j) {
        srel[wave][j + 8 * grp][lrow]      = rel0[j];
        srel[wave][j + 8 * grp][16 + lrow] = rel1[j];
    }
    const float* rrow = &srel[wave][lrow][0];
    float4 r0 = *(const float4*)(rrow + kbase);
    float4 r1 = *(const float4*)(rrow + kbase + 4);
    float4 r2 = *(const float4*)(rrow + kbase + 16);
    float4 r3 = *(const float4*)(rrow + kbase + 20);
    float ra[16] = { r0.x, r0.y, r0.z, r0.w, r1.x, r1.y, r1.z, r1.w,
                     r2.x, r2.y, r2.z, r2.w, r3.x, r3.y, r3.z, r3.w };

    float part = 0.f;
#pragma unroll
    for (int i = 0; i < 16; ++i) {
        const float t = ur[i] + ra[i] - ir[i];
        part += t * t;
    }
    part += __shfl_xor(part, 16, 32);   // combine the two K-halves

    if (lane < 16 && grow < B) out[grow] = -part;
}

extern "C" void kernel_launch(void* const* d_in, const int* in_sizes, int n_in,
                              void* d_out, int out_size, void* d_ws, size_t ws_size,
                              hipStream_t stream) {
    (void)n_in; (void)out_size; (void)d_ws; (void)ws_size;
    const int*   user_ids = (const int*)d_in[0];
    const int*   item_ids = (const int*)d_in[1];
    const float* user_emb = (const float*)d_in[2];
    const float* item_emb = (const float*)d_in[3];
    const float* W_att    = (const float*)d_in[4];
    const float* mem_tbl  = (const float*)d_in[5];
    float*       out      = (float*)d_out;

    const int B = in_sizes[0];
    const int blocks = (B + ROWS_PER_BLOCK - 1) / ROWS_PER_BLOCK;
    hipLaunchKernelGGL(lrml_fused_wmma, dim3(blocks), dim3(256), 0, stream,
                       user_ids, item_ids, user_emb, item_emb, W_att, mem_tbl, out, B);
}
